// attention_7404523618714
// MI455X (gfx1250) — matrix operand order repaired and checked by reading, not compiled, not run on hardware
//
#include <hip/hip_runtime.h>

// Problem constants (match reference)
constexpr int SEQ    = 4096;
constexpr int DIM_IN = 1024;
constexpr int DK     = 64;
constexpr int HEADS  = 16;

typedef __attribute__((ext_vector_type(16))) __bf16 v16bf;
typedef __attribute__((ext_vector_type(8)))  float  v8f;
typedef unsigned int u32x4 __attribute__((ext_vector_type(4)));
typedef int          i32x8 __attribute__((ext_vector_type(8)));
typedef int          i32x4 __attribute__((ext_vector_type(4)));

#if defined(__gfx1250__) && __has_builtin(__builtin_amdgcn_tensor_load_to_lds)
#define HAVE_TDM 1
#else
#define HAVE_TDM 0
#endif

__device__ __forceinline__ v8f wmma_bf16(v16bf a, v16bf b, v8f c) {
  return __builtin_amdgcn_wmma_f32_16x16x32_bf16(false, a, false, b, (short)0, c,
                                                 false, false);
}

// ---- CDNA5 wait helpers ----------------------------------------------------
template <int N> __device__ __forceinline__ void wait_tensorcnt() {
#if __has_builtin(__builtin_amdgcn_s_wait_tensorcnt)
  __builtin_amdgcn_s_wait_tensorcnt(N);
#else
  asm volatile("s_wait_tensorcnt %0" ::"i"(N) : "memory");
#endif
}
template <int N> __device__ __forceinline__ void wait_asynccnt() {
#if __has_builtin(__builtin_amdgcn_s_wait_asynccnt)
  __builtin_amdgcn_s_wait_asynccnt(N);
#else
  asm volatile("s_wait_asynccnt %0" ::"i"(N) : "memory");
#endif
}

// Generic (flat) __shared__ pointer -> raw LDS byte offset (low 32 bits).
__device__ __forceinline__ unsigned lds_off(const void* p) {
  return (unsigned)(size_t)p;
}

// Per-lane async copy of 16 bytes global -> LDS (GLOBAL_LOAD_ASYNC_TO_LDS_B128,
// tracked with ASYNCcnt).
__device__ __forceinline__ void async_copy_b128(unsigned lds_byte, const void* gptr) {
  unsigned long long ga = (unsigned long long)gptr;
  asm volatile("global_load_async_to_lds_b128 %0, %1, off" ::"v"(lds_byte), "v"(ga)
               : "memory");
}

#if HAVE_TDM
// One-shot TDM load of a [rows x cols] bf16 tile (row stride `ld` elements)
// into LDS.  Descriptor per CDNA5 D# layout (CDNA5 ISA).
// This toolchain uses the 6-arg builtin: (g0, g1, g2, g3, g4, cpol).
__device__ __forceinline__ void tdm_load_tile(unsigned lds_byte, const void* gptr,
                                              int rows, int cols, int ld) {
  unsigned long long ga = (unsigned long long)gptr;
  u32x4 g0;
  i32x8 g1;
  i32x4 g2 = {}, g3 = {};
  i32x8 g4 = {};
  g0[0] = 1u;                                   // count=1, user descriptor
  g0[1] = lds_byte;                             // lds_addr
  g0[2] = (unsigned)(ga & 0xffffffffu);         // global_addr[31:0]
  g0[3] = (unsigned)((ga >> 32) & 0x01ffffffu)  // global_addr[56:32]
          | 0x80000000u;                        // type=2 ("image")
  g1[0] = (1 << 16);                            // workgroup_mask=0, data_size=2B
  g1[1] = (cols & 0xffff) << 16;                // tensor_dim0[15:0]
  g1[2] = ((unsigned)cols >> 16) | ((rows & 0xffff) << 16);  // dim0 hi | tensor_dim1 lo
  g1[3] = ((unsigned)rows >> 16) | ((cols & 0xffff) << 16);  // dim1 hi | tile_dim0
  g1[4] = (rows & 0xffff);                      // tile_dim1 (tile_dim2=0)
  g1[5] = ld;                                   // tensor_dim0_stride[31:0]
  g1[6] = 0;
  g1[7] = 0;
  __builtin_amdgcn_tensor_load_to_lds(g0, g1, g2, g3, g4, 0);
}
#endif

// ---- Fragment helpers (CDNA5 ISA 7.12.2 layouts, wave32) -------------------
// A fragment 16x32 bf16: lane L holds row M=L&15; elem e -> K=((e&8)<<1)+(e&7)+8*(L>=16)
__device__ __forceinline__ v16bf load_a_bf16(const __bf16* p, int ld, int row0, int k0) {
  const int lane = threadIdx.x & 31;
  const __bf16* base = p + (size_t)(row0 + (lane & 15)) * ld + k0 + ((lane & 16) >> 1);
  v16bf a;
#pragma unroll
  for (int e = 0; e < 16; ++e) a[e] = base[((e & 8) << 1) + (e & 7)];
  return a;
}
// A fragment from a [16][32] LDS tile
__device__ __forceinline__ v16bf lds_a_frag32(const __bf16 (*tile)[32]) {
  const int lane = threadIdx.x & 31;
  const __bf16* base = &tile[lane & 15][(lane & 16) >> 1];
  v16bf a;
#pragma unroll
  for (int e = 0; e < 16; ++e) a[e] = base[((e & 8) << 1) + (e & 7)];
  return a;
}
// B fragment 32x16 from a [32][64] LDS tile: lane holds column N; elem e -> K=((e&8)<<1)+(e&7)+8*(L>=16)
__device__ __forceinline__ v16bf lds_b_frag(const __bf16 (*tile)[64], int k0, int n0) {
  const int lane = threadIdx.x & 31;
  const __bf16* base = &tile[k0 + ((lane & 16) >> 1)][n0 + (lane & 15)];
  v16bf b;
#pragma unroll
  for (int e = 0; e < 16; ++e) b[e] = base[(size_t)(((e & 8) << 1) + (e & 7)) * 64];
  return b;
}
// B fragment where B[k][n] = tile[n][k] (transposed view; contiguous in k)
__device__ __forceinline__ v16bf lds_bT_frag(const __bf16 (*tile)[64], int n0, int k0) {
  const int lane = threadIdx.x & 31;
  const __bf16* base = &tile[n0 + (lane & 15)][k0 + ((lane & 16) >> 1)];
  v16bf b;
#pragma unroll
  for (int e = 0; e < 16; ++e) b[e] = base[((e & 8) << 1) + (e & 7)];
  return b;
}

// ---- Stage 1: per-head QKV projections, fp32 in -> bf16 [H][SEQ][DK] -------
__global__ __launch_bounds__(128) void qkv_proj_kernel(
    const float* __restrict__ q, const float* __restrict__ k, const float* __restrict__ v,
    const float* __restrict__ Wq, const float* __restrict__ Wk, const float* __restrict__ Wv,
    __bf16* __restrict__ qh, __bf16* __restrict__ kh, __bf16* __restrict__ vh) {
  const int which = blockIdx.z;
  const int head  = blockIdx.y;
  const float* X = (which == 0) ? q : (which == 1) ? k : v;
  const float* W = ((which == 0) ? Wq : (which == 1) ? Wk : Wv) + (size_t)head * DIM_IN * DK;
  __bf16*      Y = ((which == 0) ? qh : (which == 1) ? kh : vh) + (size_t)head * SEQ * DK;

  __shared__ __bf16 As[16][32];
  __shared__ __bf16 Bs[32][64];

  const int row0 = blockIdx.x * 16;
  const int t    = threadIdx.x;
  const int n0   = (t >> 5) * 16;  // 4 waves cover 64 output cols
  v8f c = {};
  for (int kk = 0; kk < DIM_IN; kk += 32) {
    {  // stage A tile: 16x32 fp32 -> bf16, 4 elems/thread
      const int r = t >> 3, cl = (t & 7) * 4;
      const float* src = X + (size_t)(row0 + r) * DIM_IN + kk + cl;
#pragma unroll
      for (int j = 0; j < 4; ++j) As[r][cl + j] = (__bf16)src[j];
    }
    {  // stage B tile: 32x64 fp32 -> bf16, 16 elems/thread
      const int r = t >> 2, cl = (t & 3) * 16;
      const float* src = W + (size_t)(kk + r) * DK + cl;
#pragma unroll
      for (int j = 0; j < 16; ++j) Bs[r][cl + j] = (__bf16)src[j];
    }
    __syncthreads();
    c = wmma_bf16(lds_a_frag32(As), lds_b_frag(Bs, 0, n0), c);
    __syncthreads();
  }
  const int lane = t & 31;
  const int colN = lane & 15;
  const int rofs = (lane & 16) >> 1;
#pragma unroll
  for (int r = 0; r < 8; ++r)
    Y[(size_t)(row0 + rofs + r) * DK + n0 + colN] = (__bf16)c[r];
}

// ---- Stage 2: flash attention ----------------------------------------------
// Block = 4 waves = 64 query rows of one head; K/V 32x64 bf16 tiles are
// double-buffered in LDS, loaded with TDM (K) + async-to-LDS (V).
constexpr int NKB = SEQ / 32;
#if HAVE_TDM
constexpr int ASY_PER = 2;  // async instrs per wave per tile set (V only)
#else
constexpr int ASY_PER = 4;  // V + K both via async copies
#endif

__device__ __forceinline__ void load_kv_tiles(const __bf16* K, const __bf16* V, int key0,
                                              __bf16 (*Ksb)[DK], __bf16 (*Vsb)[DK]) {
  const int t = threadIdx.x;
  {  // V tile 32x64 bf16 = 4KB: 256 x 16B chunks over 128 threads (2 per thread)
    const unsigned vbase = lds_off(&Vsb[0][0]);
    const __bf16* g = V + (size_t)key0 * DK;
#pragma unroll
    for (int i = 0; i < 2; ++i) {
      const int chunk = t + i * 128;
      async_copy_b128(vbase + chunk * 16, g + chunk * 8);
    }
  }
#if HAVE_TDM
  if (t < 32)  // wave 0 issues one TDM descriptor for the whole K tile
    tdm_load_tile(lds_off(&Ksb[0][0]), K + (size_t)key0 * DK, 32, DK, DK);
#else
  {
    const unsigned kbase = lds_off(&Ksb[0][0]);
    const __bf16* g = K + (size_t)key0 * DK;
#pragma unroll
    for (int i = 0; i < 2; ++i) {
      const int chunk = t + i * 128;
      async_copy_b128(kbase + chunk * 16, g + chunk * 8);
    }
  }
#endif
}

__global__ __launch_bounds__(128) void flash_attn_kernel(
    const __bf16* __restrict__ qh, const __bf16* __restrict__ kh,
    const __bf16* __restrict__ vh, __bf16* __restrict__ oh) {
  const int head = blockIdx.y;
  const int wv   = threadIdx.x >> 5;
  const int lane = threadIdx.x & 31;
  const int q0   = blockIdx.x * 64 + wv * 16;
  const __bf16* Q = qh + (size_t)head * SEQ * DK;
  const __bf16* K = kh + (size_t)head * SEQ * DK;
  const __bf16* V = vh + (size_t)head * SEQ * DK;

  __shared__ __bf16 Ks[2][32][DK];    // 8KB double-buffered K tiles
  __shared__ __bf16 Vs[2][32][DK];    // 8KB double-buffered V tiles
  __shared__ __bf16 plds[4][16][32];  // per-wave P staging (C-layout -> A-layout)

  const v16bf qa0 = load_a_bf16(Q, DK, q0, 0);
  const v16bf qa1 = load_a_bf16(Q, DK, q0, 32);

  v8f o0 = {}, o1 = {}, o2 = {}, o3 = {};
  float m[8], l[8];
#pragma unroll
  for (int r = 0; r < 8; ++r) { m[r] = -3.0e38f; l[r] = 0.f; }

  const int   colN  = lane & 15;
  const int   rofs  = (lane & 16) >> 1;
  const float scale = 0.125f;  // 1/sqrt(DK)

  load_kv_tiles(K, V, 0, Ks[0], Vs[0]);  // prologue

  for (int kb = 0; kb < NKB; ++kb) {
    const int cur = kb & 1;
    if (kb + 1 < NKB) load_kv_tiles(K, V, (kb + 1) * 32, Ks[cur ^ 1], Vs[cur ^ 1]);

    // Wait for current buffer (leave next tile's transfers in flight).
#if HAVE_TDM
    if (threadIdx.x < 32) {
      if (kb + 1 < NKB) wait_tensorcnt<1>(); else wait_tensorcnt<0>();
    }
#endif
    if (kb + 1 < NKB) wait_asynccnt<ASY_PER>(); else wait_asynccnt<0>();
    __syncthreads();

    const __bf16 (*Kc)[DK] = Ks[cur];
    const __bf16 (*Vc)[DK] = Vs[cur];

    // S[16q x 32k] = Q * K^T
    v8f zf = {};
    v8f s0 = wmma_bf16(qa0, lds_bT_frag(Kc, 0, 0), zf);
    s0     = wmma_bf16(qa1, lds_bT_frag(Kc, 0, 32), s0);
    v8f s1 = wmma_bf16(qa0, lds_bT_frag(Kc, 16, 0), zf);
    s1     = wmma_bf16(qa1, lds_bT_frag(Kc, 16, 32), s1);

    float corr[8];
#pragma unroll
    for (int r = 0; r < 8; ++r) {
      const float x0 = s0[r] * scale;
      const float x1 = s1[r] * scale;
      float mx = fmaxf(x0, x1);
      mx = fmaxf(mx, __shfl_xor(mx, 1, 32));
      mx = fmaxf(mx, __shfl_xor(mx, 2, 32));
      mx = fmaxf(mx, __shfl_xor(mx, 4, 32));
      mx = fmaxf(mx, __shfl_xor(mx, 8, 32));  // row max over 16-lane half
      const float mn = fmaxf(m[r], mx);
      const float p0 = __expf(x0 - mn);
      const float p1 = __expf(x1 - mn);
      float rs = p0 + p1;
      rs += __shfl_xor(rs, 1, 32);
      rs += __shfl_xor(rs, 2, 32);
      rs += __shfl_xor(rs, 4, 32);
      rs += __shfl_xor(rs, 8, 32);  // row sum
      corr[r] = __expf(m[r] - mn);
      m[r] = mn;
      l[r] = l[r] * corr[r] + rs;
      plds[wv][rofs + r][colN]      = (__bf16)p0;
      plds[wv][rofs + r][16 + colN] = (__bf16)p1;
    }
    __syncthreads();  // order P writes (C layout) before A-layout reads

#pragma unroll
    for (int r = 0; r < 8; ++r) {
      o0[r] = o0[r] * corr[r];
      o1[r] = o1[r] * corr[r];
      o2[r] = o2[r] * corr[r];
      o3[r] = o3[r] * corr[r];
    }

    v16bf pa;  // re-read P as a 16x32 A fragment
    {
      const __bf16* base = &plds[wv][lane & 15][(lane & 16) >> 1];
#pragma unroll
      for (int e = 0; e < 16; ++e) pa[e] = base[((e & 8) << 1) + (e & 7)];
    }
    o0 = wmma_bf16(pa, lds_b_frag(Vc, 0, 0),  o0);
    o1 = wmma_bf16(pa, lds_b_frag(Vc, 0, 16), o1);
    o2 = wmma_bf16(pa, lds_b_frag(Vc, 0, 32), o2);
    o3 = wmma_bf16(pa, lds_b_frag(Vc, 0, 48), o3);

    __syncthreads();  // tiles fully consumed before next prefetch overwrites
  }

#pragma unroll
  for (int r = 0; r < 8; ++r) {
    const float inv = 1.0f / l[r];
    __bf16* dst = oh + (size_t)(q0 + rofs + r) * (HEADS * DK) + head * DK + colN;
    dst[0]  = (__bf16)(o0[r] * inv);
    dst[16] = (__bf16)(o1[r] * inv);
    dst[32] = (__bf16)(o2[r] * inv);
    dst[48] = (__bf16)(o3[r] * inv);
  }
}

// ---- Stage 3: output projection [SEQ x 1024] * Wo[1024 x 1024] -> fp32 -----
__global__ __launch_bounds__(128) void out_proj_kernel(
    const __bf16* __restrict__ oh, const float* __restrict__ Wo,
    float* __restrict__ out) {
  __shared__ __bf16 Bs[32][64];
  const int row0  = blockIdx.y * 16;
  const int nbase = blockIdx.x * 64;
  const int t     = threadIdx.x;
  const int n0    = (t >> 5) * 16;
  v8f c = {};
  for (int kk = 0; kk < DIM_IN; kk += 32) {
    {  // stage B tile: 32x64 fp32 -> bf16
      const int r = t >> 2, cl = (t & 3) * 16;
      const float* src = Wo + (size_t)(kk + r) * DIM_IN + nbase + cl;
#pragma unroll
      for (int j = 0; j < 16; ++j) Bs[r][cl + j] = (__bf16)src[j];
    }
    __syncthreads();
    v16bf a = load_a_bf16(oh, HEADS * DK, row0, kk);  // bf16 A direct from global
    c = wmma_bf16(a, lds_b_frag(Bs, 0, n0), c);
    __syncthreads();
  }
  const int lane = t & 31;
  const int colN = lane & 15;
  const int rofs = (lane & 16) >> 1;
#pragma unroll
  for (int r = 0; r < 8; ++r)
    out[(size_t)(row0 + rofs + r) * DIM_IN + nbase + n0 + colN] = c[r];
}

extern "C" void kernel_launch(void* const* d_in, const int* in_sizes, int n_in,
                              void* d_out, int out_size, void* d_ws, size_t ws_size,
                              hipStream_t stream) {
  const float* q  = (const float*)d_in[0];
  const float* k  = (const float*)d_in[1];
  const float* v  = (const float*)d_in[2];
  const float* Wq = (const float*)d_in[3];
  const float* Wk = (const float*)d_in[4];
  const float* Wv = (const float*)d_in[5];
  const float* Wo = (const float*)d_in[6];

  __bf16* qh = (__bf16*)d_ws;
  __bf16* kh = qh + (size_t)HEADS * SEQ * DK;
  __bf16* vh = kh + (size_t)HEADS * SEQ * DK;
  __bf16* oh = vh + (size_t)HEADS * SEQ * DK;  // 32 MB total in d_ws

  qkv_proj_kernel<<<dim3(SEQ / 16, HEADS, 3), 128, 0, stream>>>(
      q, k, v, Wq, Wk, Wv, qh, kh, vh);
  flash_attn_kernel<<<dim3(SEQ / 64, HEADS), 128, 0, stream>>>(qh, kh, vh, oh);
  out_proj_kernel<<<dim3(DIM_IN / 64, SEQ / 16), 128, 0, stream>>>(
      oh, Wo, (float*)d_out);
}
